// QKVBlock_18700287607595
// MI455X (gfx1250) — matrix operand order repaired and checked by reading, not compiled, not run on hardware
//
#include <hip/hip_runtime.h>

#define BB   4
#define SS   2048
#define DD   1024
#define HH   16
#define DKK  64
#define DOUTN 1024
#define NQKV (3 * HH * DKK)   // 3072

typedef __attribute__((ext_vector_type(16))) _Float16 v16h;
typedef __attribute__((ext_vector_type(8)))  float    v8f;

// ---- WMMA fragment helpers (layouts per CDNA5 ISA §7.12.2) ----

// A-matrix 16x32 f16: lane&15 = M; lanes 0-15 hold K {0..7,16..23}, lanes 16-31 hold K {8..15,24..31}
__device__ __forceinline__ int a_k_of(int e, int g) {
  return ((e >> 3) << 4) | (((e >> 1) & 3) << 1) | (g << 3) | (e & 1);
}

__device__ __forceinline__ v16h load_a_lds(const _Float16* base, int lda, int lane) {
  v16h a;
  int m = lane & 15, g = (lane >> 4) & 1;
  const _Float16* row = base + m * lda;
#pragma unroll
  for (int e = 0; e < 16; ++e) a[e] = row[a_k_of(e, g)];
  return a;
}

// B-matrix 32x16 f16 (K x N): lane&15 = N, element = K in the same order as A
__device__ __forceinline__ v16h load_b_lds(const _Float16* base, int ldb, int lane) {
  v16h b;
  int n = lane & 15, g = (lane >> 4) & 1;
#pragma unroll
  for (int e = 0; e < 16; ++e) b[e] = base[a_k_of(e, g) * ldb + n];
  return b;
}

__device__ __forceinline__ v8f wmma16(v16h a, v16h b, v8f c) {
  return __builtin_amdgcn_wmma_f32_16x16x32_f16(false, a, false, b, (short)0, c,
                                                false, false);
}

// ---- CDNA5 async memory->LDS copy (ASYNCcnt path, ISA 08 §4) ----
// 16 bytes per lane: LDS dest offset in a VGPR, 64-bit global address in a VGPR pair.
__device__ __forceinline__ void async_g2l_b128(const _Float16* gsrc, _Float16* ldst) {
  unsigned lds_off = (unsigned)(size_t)ldst;               // LDS aperture: offset in ADDR[31:0]
  unsigned long long ga = (unsigned long long)(size_t)gsrc;
  asm volatile("global_load_async_to_lds_b128 %0, %1, off"
               :: "v"(lds_off), "v"(ga) : "memory");
}
__device__ __forceinline__ void wait_async0() {
  asm volatile("s_wait_asynccnt 0" ::: "memory");
}

// ---------------- stage 0: conversions / packing ----------------

__global__ void cvt_kernel(const float* __restrict__ in, _Float16* __restrict__ o, int n) {
  int i = blockIdx.x * 256 + threadIdx.x;
  if (i < n) o[i] = (_Float16)in[i];
}

__global__ void pack_wqkv_kernel(const float* __restrict__ Wq, const float* __restrict__ Wk,
                                 const float* __restrict__ Wv, _Float16* __restrict__ w, int n) {
  int i = blockIdx.x * 256 + threadIdx.x;
  if (i >= n) return;
  const int per = HH * DD * DKK;
  int sel = i / per, rem = i % per;
  int h  = rem / (DD * DKK);
  int r2 = rem % (DD * DKK);
  int d = r2 / DKK, kk = r2 % DKK;
  const float* src = (sel == 0) ? Wq : (sel == 1) ? Wk : Wv;
  w[(size_t)d * NQKV + sel * (HH * DKK) + h * DKK + kk] = (_Float16)src[rem];
}

__global__ void pack_bias_kernel(const float* __restrict__ bq, const float* __restrict__ bk,
                                 const float* __restrict__ bv, float* __restrict__ o, int n) {
  int i = blockIdx.x * 256 + threadIdx.x;
  if (i >= n) return;
  int sel = i / (HH * DKK), j = i % (HH * DKK);
  const float* s = (sel == 0) ? bq : (sel == 1) ? bk : bv;
  o[i] = s[j];
}

// ---------------- stage 1: fused QKV projection GEMM ----------------
// 128x64 block tile, K-step 64 (2 WMMA chunks), async LDS staging.

__global__ __launch_bounds__(256) void gemm_qkv_kernel(
    const _Float16* __restrict__ xh, const _Float16* __restrict__ wqkv,
    const float* __restrict__ bqkv, _Float16* __restrict__ qkv) {
  const int NT = NQKV / 64;   // 48
  const int MT = SS / 128;    // 16
  int bz = blockIdx.x;
  int nT = bz % NT;
  int mT = (bz / NT) % MT;
  int bb = bz / (NT * MT);

  const _Float16* Ab = xh + ((size_t)bb * SS + (size_t)mT * 128) * DD;
  const _Float16* Bb = wqkv + nT * 64;

  __shared__ _Float16 As[128][72];   // 128x64 tile, pad 8 (16B-aligned rows)
  __shared__ _Float16 Bs[64][72];    // 64x64 tile

  int tid = threadIdx.x, lane = tid & 31, wv = tid >> 5;
  int wr = wv >> 1, wc = wv & 1;     // 4x2 wave grid, 32x32 per wave

  v8f acc[2][2] = {};

  for (int k0 = 0; k0 < DD; k0 += 64) {
    __syncthreads();
#pragma unroll
    for (int i = 0; i < 4; ++i) {                 // A: 128x64 halfs
      int slot = tid + i * 256;
      int row = slot >> 3, grp = slot & 7;
      async_g2l_b128(Ab + (size_t)row * DD + k0 + grp * 8, &As[row][grp * 8]);
    }
#pragma unroll
    for (int i = 0; i < 2; ++i) {                 // B: 64x64 halfs
      int slot = tid + i * 256;
      int row = slot >> 3, grp = slot & 7;
      async_g2l_b128(Bb + (size_t)(k0 + row) * NQKV + grp * 8, &Bs[row][grp * 8]);
    }
    if (k0 + 64 < DD)
      __builtin_prefetch(Ab + (size_t)(tid >> 1) * DD + k0 + 64, 0, 1);
    wait_async0();
    __syncthreads();

#pragma unroll
    for (int c = 0; c < 2; ++c) {
      v16h a0 = load_a_lds(&As[wr * 32][c * 32],      72, lane);
      v16h a1 = load_a_lds(&As[wr * 32 + 16][c * 32], 72, lane);
      v16h b0 = load_b_lds(&Bs[c * 32][wc * 32],      72, lane);
      v16h b1 = load_b_lds(&Bs[c * 32][wc * 32 + 16], 72, lane);
      acc[0][0] = wmma16(a0, b0, acc[0][0]);
      acc[0][1] = wmma16(a0, b1, acc[0][1]);
      acc[1][0] = wmma16(a1, b0, acc[1][0]);
      acc[1][1] = wmma16(a1, b1, acc[1][1]);
    }
  }

  int g = lane >> 4, nl = lane & 15;
#pragma unroll
  for (int mi = 0; mi < 2; ++mi)
#pragma unroll
    for (int ni = 0; ni < 2; ++ni)
#pragma unroll
      for (int r = 0; r < 8; ++r) {
        int m = mT * 128 + wr * 32 + mi * 16 + g * 8 + r;
        int n = nT * 64 + wc * 32 + ni * 16 + nl;
        float v = acc[mi][ni][r] + bqkv[n];
        int sel = n >> 10, h = (n >> 6) & 15, kk = n & 63;
        qkv[((((size_t)sel * BB + bb) * HH + h) * SS + m) * DKK + kk] = (_Float16)v;
      }
}

// ---------------- stage 2: flash attention ----------------
// grid: B*H*(S/64); 128 threads = 4 waves, 16 query rows each.

__global__ __launch_bounds__(128) void attn_kernel(
    const _Float16* __restrict__ qkv, _Float16* __restrict__ heads) {
  const int QT = SS / 64;  // 32
  int qt = blockIdx.x % QT;
  int bh = blockIdx.x / QT;
  int h = bh % HH, bb = bh / HH;

  const _Float16* Qb = qkv + (((size_t)0 * BB + bb) * HH + h) * SS * DKK;
  const _Float16* Kb = qkv + (((size_t)1 * BB + bb) * HH + h) * SS * DKK;
  const _Float16* Vb = qkv + (((size_t)2 * BB + bb) * HH + h) * SS * DKK;

  __shared__ _Float16 Klds[32][72];
  __shared__ _Float16 Vlds[32][72];
  __shared__ _Float16 Plds[4][16][40];

  int tid = threadIdx.x, lane = tid & 31, wv = tid >> 5;
  int g = lane >> 4, nl = lane & 15, m = lane & 15;
  int qrow = qt * 64 + wv * 16;

  // Q fragments, dk chunks [0,32) and [32,64)
  v16h qa0, qa1;
  {
    const _Float16* qp = Qb + (size_t)(qrow + m) * DKK;
#pragma unroll
    for (int e = 0; e < 16; ++e) {
      int k = a_k_of(e, g);
      qa0[e] = qp[k];
      qa1[e] = qp[32 + k];
    }
  }

  v8f oacc[4] = {};
  float mrow[8], lrow[8];
#pragma unroll
  for (int r = 0; r < 8; ++r) { mrow[r] = -3.0e38f; lrow[r] = 0.0f; }
  const float sc = 0.125f;  // 1/sqrt(64)

  for (int t = 0; t < SS; t += 32) {
    __syncthreads();
#pragma unroll
    for (int i = 0; i < 2; ++i) {                 // K,V: 32x64 halfs each
      int slot = tid + i * 128;
      int row = slot >> 3, grp = slot & 7;
      async_g2l_b128(Kb + (size_t)(t + row) * DKK + grp * 8, &Klds[row][grp * 8]);
      async_g2l_b128(Vb + (size_t)(t + row) * DKK + grp * 8, &Vlds[row][grp * 8]);
    }
    if (t + 32 < SS)
      __builtin_prefetch(Kb + (size_t)(t + 32 + (tid >> 2)) * DKK, 0, 1);
    wait_async0();
    __syncthreads();

    // scores: Q (16x64) x K^T -> two 16x16 tiles (keys t..t+15, t+16..t+31)
    v8f st[2] = {};
#pragma unroll
    for (int j = 0; j < 2; ++j) {
      v16h bk0, bk1;
#pragma unroll
      for (int e = 0; e < 16; ++e) {
        bk0[e] = Klds[j * 16 + nl][a_k_of(e, g)];        // key = nl, dk = a_k_of(e,g)
        bk1[e] = Klds[j * 16 + nl][32 + a_k_of(e, g)];   // key = nl, dk = 32+a_k_of(e,g)
      }
      st[j] = wmma16(qa0, bk0, st[j]);
      st[j] = wmma16(qa1, bk1, st[j]);
    }

    // online softmax over this 32-key block; row = g*8+r, col = nl
#pragma unroll
    for (int r = 0; r < 8; ++r) {
      float s0 = sc * st[0][r];
      float s1 = sc * st[1][r];
      float mx = fmaxf(s0, s1);
      mx = fmaxf(mx, __shfl_xor(mx, 1, 32));
      mx = fmaxf(mx, __shfl_xor(mx, 2, 32));
      mx = fmaxf(mx, __shfl_xor(mx, 4, 32));
      mx = fmaxf(mx, __shfl_xor(mx, 8, 32));
      float mn = fmaxf(mrow[r], mx);
      float alpha = __expf(mrow[r] - mn);
      float p0 = __expf(s0 - mn);
      float p1 = __expf(s1 - mn);
      float sum = p0 + p1;
      sum += __shfl_xor(sum, 1, 32);
      sum += __shfl_xor(sum, 2, 32);
      sum += __shfl_xor(sum, 4, 32);
      sum += __shfl_xor(sum, 8, 32);
      lrow[r] = lrow[r] * alpha + sum;
      mrow[r] = mn;
#pragma unroll
      for (int j = 0; j < 4; ++j) oacc[j][r] *= alpha;
      Plds[wv][g * 8 + r][nl] = (_Float16)p0;        // key t+nl
      Plds[wv][g * 8 + r][16 + nl] = (_Float16)p1;   // key t+16+nl
    }
    asm volatile("s_wait_dscnt 0" ::: "memory");  // P stores -> A-frag loads (same wave)

    v16h pa = load_a_lds(&Plds[wv][0][0], 40, lane);
#pragma unroll
    for (int j = 0; j < 4; ++j) {
      v16h bv_ = load_b_lds(&Vlds[0][j * 16], 72, lane);  // lane = dk, e = key
      oacc[j] = wmma16(pa, bv_, oacc[j]);
    }
  }

  // normalize and store (concat-head layout [B][S][H*DK])
#pragma unroll
  for (int r = 0; r < 8; ++r) {
    float inv = 1.0f / lrow[r];
    int s_idx = qrow + g * 8 + r;
#pragma unroll
    for (int j = 0; j < 4; ++j) {
      float v = oacc[j][r] * inv;
      heads[((size_t)bb * SS + s_idx) * (HH * DKK) + h * DKK + j * 16 + nl] = (_Float16)v;
    }
  }
}

// ---------------- stage 3: output projection GEMM ----------------

__global__ __launch_bounds__(256) void gemm_out_kernel(
    const _Float16* __restrict__ heads, const _Float16* __restrict__ woh,
    const float* __restrict__ bo, float* __restrict__ out) {
  const int NT = DOUTN / 64;  // 16
  int nT = blockIdx.x % NT;
  int mT = blockIdx.x / NT;   // 0..63
  const int KD = HH * DKK;    // 1024
  const _Float16* Ab = heads + (size_t)mT * 128 * KD;
  const _Float16* Bb = woh + nT * 64;

  __shared__ _Float16 As[128][72];
  __shared__ _Float16 Bs[64][72];

  int tid = threadIdx.x, lane = tid & 31, wv = tid >> 5;
  int wr = wv >> 1, wc = wv & 1;

  v8f acc[2][2] = {};

  for (int k0 = 0; k0 < KD; k0 += 64) {
    __syncthreads();
#pragma unroll
    for (int i = 0; i < 4; ++i) {
      int slot = tid + i * 256;
      int row = slot >> 3, grp = slot & 7;
      async_g2l_b128(Ab + (size_t)row * KD + k0 + grp * 8, &As[row][grp * 8]);
    }
#pragma unroll
    for (int i = 0; i < 2; ++i) {
      int slot = tid + i * 256;
      int row = slot >> 3, grp = slot & 7;
      async_g2l_b128(Bb + (size_t)(k0 + row) * DOUTN + grp * 8, &Bs[row][grp * 8]);
    }
    if (k0 + 64 < KD)
      __builtin_prefetch(Ab + (size_t)(tid >> 1) * KD + k0 + 64, 0, 1);
    wait_async0();
    __syncthreads();

#pragma unroll
    for (int c = 0; c < 2; ++c) {
      v16h a0 = load_a_lds(&As[wr * 32][c * 32],      72, lane);
      v16h a1 = load_a_lds(&As[wr * 32 + 16][c * 32], 72, lane);
      v16h b0 = load_b_lds(&Bs[c * 32][wc * 32],      72, lane);
      v16h b1 = load_b_lds(&Bs[c * 32][wc * 32 + 16], 72, lane);
      acc[0][0] = wmma16(a0, b0, acc[0][0]);
      acc[0][1] = wmma16(a0, b1, acc[0][1]);
      acc[1][0] = wmma16(a1, b0, acc[1][0]);
      acc[1][1] = wmma16(a1, b1, acc[1][1]);
    }
  }

  int g = lane >> 4, nl = lane & 15;
#pragma unroll
  for (int mi = 0; mi < 2; ++mi)
#pragma unroll
    for (int ni = 0; ni < 2; ++ni)
#pragma unroll
      for (int r = 0; r < 8; ++r) {
        int mr = mT * 128 + wr * 32 + mi * 16 + g * 8 + r;
        int n = nT * 64 + wc * 32 + ni * 16 + nl;
        out[(size_t)mr * DOUTN + n] = acc[mi][ni][r] + bo[n];
      }
}

// ---------------- launcher ----------------

extern "C" void kernel_launch(void* const* d_in, const int* in_sizes, int n_in,
                              void* d_out, int out_size, void* d_ws, size_t ws_size,
                              hipStream_t stream) {
  const float* x  = (const float*)d_in[0];
  const float* Wq = (const float*)d_in[1];
  const float* bq = (const float*)d_in[2];
  const float* Wk = (const float*)d_in[3];
  const float* bk = (const float*)d_in[4];
  const float* Wv = (const float*)d_in[5];
  const float* bv = (const float*)d_in[6];
  const float* Wo = (const float*)d_in[7];
  const float* bo = (const float*)d_in[8];
  float* out = (float*)d_out;

  auto align256 = [](size_t v) { return (v + 255) & ~(size_t)255; };
  char* p = (char*)d_ws;
  _Float16* xh    = (_Float16*)p; p += align256((size_t)BB * SS * DD * 2);        // 16 MB
  _Float16* wqkv  = (_Float16*)p; p += align256((size_t)DD * NQKV * 2);           // 6 MB
  float*    bqkv  = (float*)p;    p += align256((size_t)NQKV * 4);
  _Float16* woh   = (_Float16*)p; p += align256((size_t)HH * DKK * DOUTN * 2);    // 2 MB
  _Float16* qkvb  = (_Float16*)p; p += align256((size_t)3 * BB * HH * SS * DKK * 2); // 50 MB
  _Float16* headsb = (_Float16*)p;                                                // 16 MB

  { int n = BB * SS * DD;     cvt_kernel<<<(n + 255) / 256, 256, 0, stream>>>(x, xh, n); }
  { int n = HH * DKK * DOUTN; cvt_kernel<<<(n + 255) / 256, 256, 0, stream>>>(Wo, woh, n); }
  { int n = 3 * HH * DD * DKK; pack_wqkv_kernel<<<(n + 255) / 256, 256, 0, stream>>>(Wq, Wk, Wv, wqkv, n); }
  { int n = NQKV;             pack_bias_kernel<<<(n + 255) / 256, 256, 0, stream>>>(bq, bk, bv, bqkv, n); }

  gemm_qkv_kernel<<<BB * (SS / 128) * (NQKV / 64), 256, 0, stream>>>(xh, wqkv, bqkv, qkvb);
  attn_kernel<<<BB * HH * (SS / 64), 128, 0, stream>>>(qkvb, headsb);
  gemm_out_kernel<<<(BB * SS / 128) * (DOUTN / 64), 256, 0, stream>>>(headsb, woh, bo, out);
}
